// DMRI2INetworkLayer_26508538150958
// MI455X (gfx1250) — matrix operand order repaired and checked by reading, not compiled, not run on hardware
//
#include <hip/hip_runtime.h>
#include <hip/hip_bf16.h>

// Problem dims
#define B_SZ 4096
#define S_SZ 200
#define D_SZ 128
#define H_SZ 64
#define SP   208   // S padded to multiple of 16
#define MT   13    // SP/16 m-tiles
#define NT   4     // H/16  n-tiles
#define KT   4     // D/32  k-steps

typedef __attribute__((ext_vector_type(16))) __bf16 bf16x16;
typedef __attribute__((ext_vector_type(8)))  float  f32x8;

union Frag {
    uint4   u[2];
    bf16x16 v;
};

union PackBF4 {
    __bf16 h[4];
    uint2  u;
};

// branch-free tanh: 1 - 2/(exp(2x)+1); saturates to +-1, NaN-propagating.
__device__ __forceinline__ float fast_tanh(float x) {
    float e2 = __expf(2.0f * x);                       // v_exp_f32 path
    return 1.0f - 2.0f * __builtin_amdgcn_rcpf(e2 + 1.0f);
}

// DPP16 row-rotate + add (stays within each 16-lane row on wave32).
// After ror8/ror4/ror2/ror1 steps, every lane holds the 16-lane row sum.
#define DPP_ROR_ADD(c, CTRL)                                                    \
    do {                                                                        \
        int _ci = __builtin_bit_cast(int, (c));                                 \
        int _sh = __builtin_amdgcn_update_dpp(0, _ci, (CTRL), 0xF, 0xF, true);  \
        (c) += __builtin_bit_cast(float, _sh);                                  \
    } while (0)

__device__ __forceinline__ float row16_reduce_add(float c) {
    DPP_ROR_ADD(c, 0x128);   // row_ror:8
    DPP_ROR_ADD(c, 0x124);   // row_ror:4
    DPP_ROR_ADD(c, 0x122);   // row_ror:2
    DPP_ROR_ADD(c, 0x121);   // row_ror:1
    return c;
}

// ---------- prep: We[k][n] -> WeT bf16 [n][k] ----------
__global__ void k_prep_weT(const float* __restrict__ We, __bf16* __restrict__ WeT) {
    int i = blockIdx.x * 256 + threadIdx.x;       // 0..8191
    int n = i >> 7, k = i & 127;
    WeT[i] = (__bf16)We[k * H_SZ + n];
}

// ---------- prep: P[s][h] = pos[s,:] . Wp[:,h] ----------
__global__ void k_prep_P(const float* __restrict__ pos, const float* __restrict__ Wp,
                         float* __restrict__ P) {
    int i = blockIdx.x * 256 + threadIdx.x;
    if (i >= S_SZ * H_SZ) return;
    int s = i >> 6, h = i & 63;
    float acc = 0.f;
#pragma unroll 8
    for (int k = 0; k < D_SZ; ++k)
        acc += pos[s * D_SZ + k] * Wp[k * H_SZ + h];
    P[i] = acc;
}

// ---------- main fused kernel: one block per batch row ----------
__launch_bounds__(256, 1)
__global__ void k_main(const float* __restrict__ Xs,        // [B,S,D]
                       const float* __restrict__ Xitem,     // [B,D]
                       const unsigned char* __restrict__ mask, // [B,S] bool
                       const float* __restrict__ Wc,        // [D,H]
                       const float* __restrict__ z,         // [H]
                       const __bf16* __restrict__ WeT,      // [H][D] bf16
                       const float* __restrict__ P,         // [S][H]
                       float* __restrict__ out)             // [B, D+1]
{
    __shared__ __attribute__((aligned(16))) __bf16 sXs[SP * D_SZ];     // 53248 B
    __shared__ __attribute__((aligned(16))) __bf16 sWeT[H_SZ * D_SZ];  // 16384 B
    __shared__ __attribute__((aligned(16))) float  sP[SP * H_SZ];      // 53248 B
    __shared__ float sXc[H_SZ];
    __shared__ float sZ[H_SZ];
    __shared__ __attribute__((aligned(16))) float sXitem[D_SZ];
    __shared__ float sScores[SP];
    __shared__ float sAttn[SP];
    __shared__ float sRed[256];
    __shared__ float sScal[2];

    const int tid = threadIdx.x;
    const int b   = blockIdx.x;
    const float* xsrow = Xs + (size_t)b * (S_SZ * D_SZ);

    // ---------------- phase 0: cooperative loads ----------------
    // X_series row -> bf16 LDS (6400 float4, 25 per thread)
    {
        const float4* src = (const float4*)xsrow;
#pragma unroll 5
        for (int j = 0; j < 25; ++j) {
            int i = tid + 256 * j;              // float4 index
            float4 f = src[i];
            PackBF4 pk;
            pk.h[0] = (__bf16)f.x; pk.h[1] = (__bf16)f.y;
            pk.h[2] = (__bf16)f.z; pk.h[3] = (__bf16)f.w;
            *(uint2*)&sXs[i * 4] = pk.u;
        }
        // zero pad rows 200..207 (1024 bf16 = 256 uint2)
        uint2 zz; zz.x = 0u; zz.y = 0u;
        ((uint2*)&sXs[S_SZ * D_SZ])[tid] = zz;
    }
    // WeT -> LDS (1024 uint4)
    {
        const uint4* src = (const uint4*)WeT;
        uint4* dst = (uint4*)sWeT;
#pragma unroll
        for (int j = 0; j < 4; ++j) dst[tid + 256 * j] = src[tid + 256 * j];
    }
    // P -> LDS (3200 uint4)
    {
        const uint4* src = (const uint4*)P;
        uint4* dst = (uint4*)sP;
        for (int i = tid; i < (S_SZ * H_SZ) / 4; i += 256) dst[i] = src[i];
    }
    if (tid < H_SZ) sZ[tid] = z[tid];
    if (tid < 32)   ((float4*)sXitem)[tid] = ((const float4*)(Xitem + (size_t)b * D_SZ))[tid];
    if (tid < SP)   sScores[tid] = 0.f;
    __syncthreads();

    // Xc[n] = X_item[b,:] . Wc[:,n]  (threads 0..63, coalesced over n per k)
    if (tid < H_SZ) {
        float acc = 0.f;
#pragma unroll 16
        for (int k = 0; k < D_SZ; ++k) acc += sXitem[k] * Wc[k * H_SZ + tid];
        sXc[tid] = acc;
    }
    __syncthreads();

    // ---------------- phase 1: WMMA GEMM + tanh + z-dot ----------------
    const int wave = tid >> 5, lane = tid & 31;
    const int lrow = lane & 15, lhi = lane >> 4;

    for (int p = wave; p < MT * NT; p += 8) {
        const int m0 = (p % MT) * 16;
        const int n0 = (p / MT) * 16;
        f32x8 acc = {};
#pragma unroll
        for (int kt = 0; kt < KT; ++kt) {
            // A: 16x32 bf16 tile of X_series (rows m0..m0+15)
            Frag a;
            const int ae = (m0 + lrow) * D_SZ + kt * 32 + lhi * 8;
            a.u[0] = *(const uint4*)&sXs[ae];        // K 0..7  (or 8..15)
            a.u[1] = *(const uint4*)&sXs[ae + 16];   // K 16..23(or 24..31)
            // B: 32x16 bf16 tile of We (cols n0..n0+15), column per lane
            Frag bb;
            const int be = (n0 + lrow) * D_SZ + kt * 32 + lhi * 8;
            bb.u[0] = *(const uint4*)&sWeT[be];      // K 0..7  (or 8..15)
            bb.u[1] = *(const uint4*)&sWeT[be + 16]; // K 16..23(or 24..31)
            acc = __builtin_amdgcn_wmma_f32_16x16x32_bf16(
                false, a.v, false, bb.v, (short)0, acc, false, false);
        }
        const int n   = n0 + lrow;
        const float zn  = sZ[n];
        const float xcn = sXc[n];
#pragma unroll
        for (int r = 0; r < 8; ++r) {
            const int m = m0 + r + lhi * 8;
            float v = acc[r] + xcn + sP[m * H_SZ + n];
            float c = fast_tanh(v) * zn;
            // pure-VALU DPP reduction over the 16 n-values of row m
            c = row16_reduce_add(c);
            if (lrow == 0) atomicAdd(&sScores[m], c);   // non-returning ds_add_f32
        }
    }
    __syncthreads();

    // Pull this thread's phase-3 stream back toward the WGP while softmax runs.
    {
        const int d0 = tid & 127, h0 = tid >> 7;
        __builtin_prefetch(&xsrow[h0 * D_SZ + d0], 0, 3);   // global_prefetch_b8
    }

    // ---------------- phase 2: masked softmax + raw score sum ----------------
    float raw = 0.f, maskedRaw = 0.f, sc = -__builtin_inff();
    if (tid < S_SZ) {
        raw = sScores[tid];
        const bool vm = mask[(size_t)b * S_SZ + tid] != 0;
        sc        = vm ? raw : -__builtin_inff();
        maskedRaw = vm ? raw : 0.f;
    }
    sRed[tid] = sc; __syncthreads();
#pragma unroll
    for (int s2 = 128; s2 > 0; s2 >>= 1) {
        if (tid < s2) sRed[tid] = fmaxf(sRed[tid], sRed[tid + s2]);
        __syncthreads();
    }
    const float mx = sRed[0];
    __syncthreads();

    const float e = (tid < S_SZ) ? __expf(sc - mx) : 0.f;   // exp(-inf)=0 for masked
    if (tid < S_SZ) sAttn[tid] = e;

    sRed[tid] = e; __syncthreads();
#pragma unroll
    for (int s2 = 128; s2 > 0; s2 >>= 1) {
        if (tid < s2) sRed[tid] += sRed[tid + s2];
        __syncthreads();
    }
    if (tid == 0) sScal[0] = __builtin_amdgcn_rcpf(sRed[0]);
    __syncthreads();

    sRed[tid] = maskedRaw; __syncthreads();
#pragma unroll
    for (int s2 = 128; s2 > 0; s2 >>= 1) {
        if (tid < s2) sRed[tid] += sRed[tid + s2];
        __syncthreads();
    }
    if (tid == 0) sScal[1] = sRed[0];
    __syncthreads();
    const float invSum   = sScal[0];
    const float scoreSum = sScal[1];

    // ---------------- phase 3: attn-weighted sum over S (fp32 from global/L2) ----
    const int d  = tid & 127;
    const int hh = tid >> 7;
    float po = 0.f;
    for (int s = hh; s < S_SZ; s += 2)
        po += sAttn[s] * xsrow[s * D_SZ + d];
    sRed[tid] = po; __syncthreads();
    if (tid < 128)
        out[(size_t)b * (D_SZ + 1) + tid] = (sRed[tid] + sRed[tid + 128]) * invSum;
    if (tid == 255)
        out[(size_t)b * (D_SZ + 1) + D_SZ] = scoreSum;
}

extern "C" void kernel_launch(void* const* d_in, const int* in_sizes, int n_in,
                              void* d_out, int out_size, void* d_ws, size_t ws_size,
                              hipStream_t stream) {
    (void)in_sizes; (void)n_in; (void)out_size; (void)ws_size;
    const float*         X_series = (const float*)d_in[0];
    const float*         pos      = (const float*)d_in[1];
    const float*         X_item   = (const float*)d_in[2];
    const unsigned char* vmask    = (const unsigned char*)d_in[3];
    const float*         Wc       = (const float*)d_in[4];
    const float*         Wp       = (const float*)d_in[5];
    const float*         We       = (const float*)d_in[6];
    const float*         z        = (const float*)d_in[7];
    float* out = (float*)d_out;

    __bf16* WeT = (__bf16*)d_ws;                      // 16384 bytes
    float*  P   = (float*)((char*)d_ws + 16384);      // 51200 bytes

    k_prep_weT<<<32, 256, 0, stream>>>(We, WeT);
    k_prep_P  <<<50, 256, 0, stream>>>(pos, Wp, P);
    k_main    <<<B_SZ, 256, 0, stream>>>(X_series, X_item, vmask, Wc, z, WeT, P, out);
}
